// GINModelBatchesExp_61014305407332
// MI455X (gfx1250) — compile-verified, matrix operand order checked by reading, not run on hardware
//
#include <hip/hip_runtime.h>

// ---------------------------------------------------------------------------
// GIN model on MI455X (gfx1250, wave32, WMMA).
//
// Roofline: ~15 GFLOP of GEMM (negligible vs WMMA peak) vs ~2.5 GB of edge
// gather/atomic traffic. agg tables (51 MB) fit in the 192 MB L2, so the
// scatter phase is L2-atomic bound; HBM sees only streaming passes.
// GEMMs use v_wmma_f32_16x16x32_f16 (f16 in, f32 accumulate); weights are
// pre-packed into the per-lane WMMA B-fragment layout so the inner loop is
// one 32-byte vector load per fragment.
// ---------------------------------------------------------------------------

typedef __attribute__((ext_vector_type(16))) _Float16 v16h;
typedef __attribute__((ext_vector_type(8)))  float    v8f;

#define F_IN 64
#define HID  128
#define NCLS 10

// --- WMMA fragment helpers --------------------------------------------------
// 16-bit A-matrix 16x32 layout (ISA 7.12.2): lane l<16 holds row M=l with
// K = {k0..k0+7, k0+16..k0+23}; lane l>=16 holds row M=l-16 with
// K = {k0+8..k0+15, k0+24..k0+31}.  B (32x16) assumed symmetric with N=lane&15.

__device__ __forceinline__ v16h load_bfrag(const _Float16* __restrict__ pk,
                                           int frag, int lane) {
  // packed layout: [frag][lane][16 halves] -> one contiguous 32B load
  return *reinterpret_cast<const v16h*>(pk + ((size_t)frag * 32 + lane) * 16);
}

__device__ __forceinline__ v16h load_afrag_g(const float* __restrict__ A, int ld,
                                             int row, int k0, int lane) {
  const float* p = A + (size_t)row * ld + k0 + ((lane >= 16) ? 8 : 0);
  v16h a;
#pragma unroll
  for (int j = 0; j < 8; ++j) {
    a[j]     = (_Float16)p[j];
    a[j + 8] = (_Float16)p[j + 16];
  }
  return a;
}

__device__ __forceinline__ v16h load_afrag_l(const _Float16* __restrict__ t,
                                             int row, int k0, int lane) {
  const _Float16* p = t + row * HID + k0 + ((lane >= 16) ? 8 : 0);
  v16h a;
#pragma unroll
  for (int j = 0; j < 8; ++j) {
    a[j]     = p[j];
    a[j + 8] = p[j + 16];
  }
  return a;
}

// --- weight packing: W (K x Nreal, row-major f32) -> B-fragment layout ------
__global__ void pack_weights_kernel(const float* __restrict__ W,
                                    _Float16* __restrict__ dst,
                                    int K, int Nreal, int KT) {
  const int f  = blockIdx.x;       // f = nt*KT + kt
  const int kt = f % KT;
  const int nt = f / KT;
  const int l  = threadIdx.x;      // 0..31
  const int ncol  = nt * 16 + (l & 15);
  const int kbase = kt * 32 + ((l >= 16) ? 8 : 0);
  v16h v;
#pragma unroll
  for (int j = 0; j < 16; ++j) {
    const int k = kbase + (j < 8 ? j : j + 8);
    float val = (ncol < Nreal && k < K) ? W[(size_t)k * Nreal + ncol] : 0.0f;
    v[j] = (_Float16)val;
  }
  *reinterpret_cast<v16h*>(dst + ((size_t)f * 32 + l) * 16) = v;
}

// --- edge scatter: agg[dst] += feat[src], lane-per-column, L2 atomics -------
__global__ __launch_bounds__(256)
void scatter_add_kernel(const float* __restrict__ feat,
                        const int* __restrict__ srcIdx,
                        const int* __restrict__ dstIdx,
                        float* __restrict__ agg, int nedges, int fdim) {
  const int t = blockIdx.x * 256 + threadIdx.x;
  const int e = t >> 5;
  if (e >= nedges) return;
  const int lane = t & 31;
  const int s = srcIdx[e];
  const int d = dstIdx[e];
  const float* xs = feat + (size_t)s * fdim;
  float* ad = agg + (size_t)d * fdim;
  for (int c = lane; c < fdim; c += 32)
    atomicAdd(&ad[c], xs[c]);   // global_atomic_add_f32, hits in L2 (51MB < 192MB)
}

// --- fused 2-layer MLP: Out = relu(A@Wa+ba) @ Wb + bb -----------------------
// 8 waves/block, each wave owns one 16-row M tile; intermediate staged in LDS.
template <int K1>
__global__ __launch_bounds__(256)
void mlp2_kernel(const float* __restrict__ A, const _Float16* __restrict__ Wa,
                 const float* __restrict__ ba, const _Float16* __restrict__ Wb,
                 const float* __restrict__ bb, float* __restrict__ Out,
                 int ntiles) {
  constexpr int KT1 = K1 / 32;
  __shared__ _Float16 t1[8][16 * HID];           // 32 KB / block
  const int lane = threadIdx.x & 31;
  const int wv   = threadIdx.x >> 5;
  const int tile = blockIdx.x * 8 + wv;
  const bool active = tile < ntiles;
  const int base = tile * 16;
  const int col  = lane & 15;
  const int r0   = (lane < 16) ? 0 : 8;

  if (active) {
    v16h af[KT1];
#pragma unroll
    for (int kt = 0; kt < KT1; ++kt)
      af[kt] = load_afrag_g(A, K1, base + col, kt * 32, lane);
#pragma unroll
    for (int nt = 0; nt < 8; ++nt) {
      v8f c = 0.0f;
#pragma unroll
      for (int kt = 0; kt < KT1; ++kt)
        c = __builtin_amdgcn_wmma_f32_16x16x32_f16(
            false, af[kt], false, load_bfrag(Wa, nt * KT1 + kt, lane),
            (short)0, c, false, false);
      const float bv = ba[nt * 16 + col];
#pragma unroll
      for (int g = 0; g < 8; ++g) {
        float v = c[g] + bv;
        t1[wv][(g + r0) * HID + nt * 16 + col] = (_Float16)(v > 0.f ? v : 0.f);
      }
    }
  }
  __syncthreads();
  if (active) {
    v16h af2[4];
#pragma unroll
    for (int kt = 0; kt < 4; ++kt)
      af2[kt] = load_afrag_l(t1[wv], col, kt * 32, lane);
#pragma unroll
    for (int nt = 0; nt < 8; ++nt) {
      v8f c = 0.0f;
#pragma unroll
      for (int kt = 0; kt < 4; ++kt)
        c = __builtin_amdgcn_wmma_f32_16x16x32_f16(
            false, af2[kt], false, load_bfrag(Wb, nt * 4 + kt, lane),
            (short)0, c, false, false);
      const float bv = bb[nt * 16 + col];
#pragma unroll
      for (int g = 0; g < 8; ++g)
        Out[(size_t)(base + g + r0) * HID + nt * 16 + col] = c[g] + bv;
    }
  }
}

// --- fused classifier: out = relu(L@W1+b1) @ W2 + b2 (N padded 10->16) ------
__global__ __launch_bounds__(256)
void classifier_kernel(const float* __restrict__ L, const _Float16* __restrict__ W1,
                       const float* __restrict__ b1, const _Float16* __restrict__ W2,
                       const float* __restrict__ b2, float* __restrict__ Out,
                       int ntiles) {
  __shared__ _Float16 t1[8][16 * HID];
  const int lane = threadIdx.x & 31;
  const int wv   = threadIdx.x >> 5;
  const int tile = blockIdx.x * 8 + wv;
  const bool active = tile < ntiles;
  const int base = tile * 16;
  const int col  = lane & 15;
  const int r0   = (lane < 16) ? 0 : 8;

  if (active) {
    v16h af[4];
#pragma unroll
    for (int kt = 0; kt < 4; ++kt)
      af[kt] = load_afrag_g(L, HID, base + col, kt * 32, lane);
#pragma unroll
    for (int nt = 0; nt < 8; ++nt) {
      v8f c = 0.0f;
#pragma unroll
      for (int kt = 0; kt < 4; ++kt)
        c = __builtin_amdgcn_wmma_f32_16x16x32_f16(
            false, af[kt], false, load_bfrag(W1, nt * 4 + kt, lane),
            (short)0, c, false, false);
      const float bv = b1[nt * 16 + col];
#pragma unroll
      for (int g = 0; g < 8; ++g) {
        float v = c[g] + bv;
        t1[wv][(g + r0) * HID + nt * 16 + col] = (_Float16)(v > 0.f ? v : 0.f);
      }
    }
  }
  __syncthreads();
  if (active) {
    v16h af2[4];
#pragma unroll
    for (int kt = 0; kt < 4; ++kt)
      af2[kt] = load_afrag_l(t1[wv], col, kt * 32, lane);
    v8f c = 0.0f;
#pragma unroll
    for (int kt = 0; kt < 4; ++kt)
      c = __builtin_amdgcn_wmma_f32_16x16x32_f16(
          false, af2[kt], false, load_bfrag(W2, kt, lane),
          (short)0, c, false, false);
    if (col < NCLS) {
      const float bv = b2[col];
#pragma unroll
      for (int g = 0; g < 8; ++g)
        Out[(size_t)(base + g + r0) * NCLS + col] = c[g] + bv;
    }
  }
}

// --- BatchNorm: column sums / sum-of-squares, then normalize+ReLU -----------
__global__ __launch_bounds__(256)
void bn_stats_kernel(const float* __restrict__ h, float* __restrict__ s1,
                     float* __restrict__ s2, int nrows) {
  const int col = threadIdx.x & (HID - 1);
  const int sub = threadIdx.x >> 7;  // 0 or 1
  float a = 0.f, b = 0.f;
  for (int r = blockIdx.x * 2 + sub; r < nrows; r += gridDim.x * 2) {
    const float v = h[(size_t)r * HID + col];
    a += v;
    b += v * v;
  }
  __shared__ float sA[256], sB[256];
  sA[threadIdx.x] = a;
  sB[threadIdx.x] = b;
  __syncthreads();
  if (threadIdx.x < HID) {
    atomicAdd(&s1[col], sA[threadIdx.x] + sA[threadIdx.x + HID]);
    atomicAdd(&s2[col], sB[threadIdx.x] + sB[threadIdx.x + HID]);
  }
}

__global__ __launch_bounds__(256)
void bn_apply_kernel(float* __restrict__ h, const float* __restrict__ s1,
                     const float* __restrict__ s2, const float* __restrict__ g,
                     const float* __restrict__ b, int total, float invN) {
  const int i = blockIdx.x * 256 + threadIdx.x;
  if (i >= total) return;
  const int col = i & (HID - 1);
  const float mean = s1[col] * invN;
  const float var  = s2[col] * invN - mean * mean;
  const float y = (h[i] - mean) * rsqrtf(var + 1e-5f) * g[col] + b[col];
  h[i] = y > 0.f ? y : 0.f;
}

__global__ void zero_kernel(float* __restrict__ p, int n) {
  const int i = blockIdx.x * blockDim.x + threadIdx.x;
  if (i < n) p[i] = 0.f;
}

// ---------------------------------------------------------------------------
extern "C" void kernel_launch(void* const* d_in, const int* in_sizes, int n_in,
                              void* d_out, int out_size, void* d_ws, size_t ws_size,
                              hipStream_t stream) {
  const float* x    = (const float*)d_in[0];
  const int*   ei   = (const int*)d_in[1];
  const float* w1a  = (const float*)d_in[2];
  const float* b1a  = (const float*)d_in[3];
  const float* w1b  = (const float*)d_in[4];
  const float* b1b  = (const float*)d_in[5];
  const float* w2a  = (const float*)d_in[6];
  const float* b2a  = (const float*)d_in[7];
  const float* w2b  = (const float*)d_in[8];
  const float* b2b  = (const float*)d_in[9];
  const float* bn1g = (const float*)d_in[10];
  const float* bn1b = (const float*)d_in[11];
  const float* bn2g = (const float*)d_in[12];
  const float* bn2b = (const float*)d_in[13];
  const float* mw1  = (const float*)d_in[14];
  const float* mb1  = (const float*)d_in[15];
  const float* mw2  = (const float*)d_in[16];
  const float* mb2  = (const float*)d_in[17];

  const int N = in_sizes[0] / F_IN;   // 100000 (divisible by 16)
  const int E = in_sizes[1] / 2;      // 1.6M
  const int* srcIdx = ei;
  const int* dstIdx = ei + E;

  float* latent = (float*)d_out;                    // N x HID
  float* clsout = latent + (size_t)N * HID;         // N x NCLS

  // workspace carve (256B aligned)
  char* ws = (char*)d_ws;
  size_t off = 0;
  auto carve = [&](size_t bytes) -> void* {
    void* p = ws + off;
    off = (off + bytes + 255) & ~(size_t)255;
    return p;
  };
  float* bufA  = (float*)carve((size_t)N * HID * sizeof(float));  // agg buffer
  float* bufB  = (float*)carve((size_t)N * HID * sizeof(float));  // h buffer
  float* stats = (float*)carve(4 * HID * sizeof(float));          // s1a,s2a,s1b,s2b
  _Float16* pk_w1a = (_Float16*)carve((size_t)2 * 8 * 32 * 16 * 2);
  _Float16* pk_w1b = (_Float16*)carve((size_t)4 * 8 * 32 * 16 * 2);
  _Float16* pk_w2a = (_Float16*)carve((size_t)4 * 8 * 32 * 16 * 2);
  _Float16* pk_w2b = (_Float16*)carve((size_t)4 * 8 * 32 * 16 * 2);
  _Float16* pk_mw1 = (_Float16*)carve((size_t)4 * 8 * 32 * 16 * 2);
  _Float16* pk_mw2 = (_Float16*)carve((size_t)4 * 1 * 32 * 16 * 2);
  (void)ws_size; (void)n_in; (void)out_size;

  // 0. zero BN stats
  zero_kernel<<<2, 256, 0, stream>>>(stats, 4 * HID);

  // 1. pack weights into WMMA B-fragment layout (tiny, once per launch)
  pack_weights_kernel<<<8 * 2, 32, 0, stream>>>(w1a, pk_w1a, F_IN, HID, 2);
  pack_weights_kernel<<<8 * 4, 32, 0, stream>>>(w1b, pk_w1b, HID, HID, 4);
  pack_weights_kernel<<<8 * 4, 32, 0, stream>>>(w2a, pk_w2a, HID, HID, 4);
  pack_weights_kernel<<<8 * 4, 32, 0, stream>>>(w2b, pk_w2b, HID, HID, 4);
  pack_weights_kernel<<<8 * 4, 32, 0, stream>>>(mw1, pk_mw1, HID, HID, 4);
  pack_weights_kernel<<<1 * 4, 32, 0, stream>>>(mw2, pk_mw2, HID, NCLS, 4);

  const int tiles = (N + 15) / 16;            // 6250
  const int mgrid = (tiles + 7) / 8;          // 8 M-tiles / block
  const int sgrid = (int)(((long long)E * 32 + 255) / 256);
  const int egrid = (int)(((size_t)N * HID + 255) / 256);
  const float invN = 1.0f / (float)N;

  // 2. conv1: agg = x + scatter_sum(x)
  hipMemcpyAsync(bufA, x, (size_t)N * F_IN * sizeof(float),
                 hipMemcpyDeviceToDevice, stream);
  scatter_add_kernel<<<sgrid, 256, 0, stream>>>(x, srcIdx, dstIdx, bufA, E, F_IN);
  // 3. h1 = mlp1(agg)
  mlp2_kernel<F_IN><<<mgrid, 256, 0, stream>>>(bufA, pk_w1a, b1a, pk_w1b, b1b,
                                               bufB, tiles);
  // 4. h1 = relu(bn1(h1)) in place
  bn_stats_kernel<<<512, 256, 0, stream>>>(bufB, stats, stats + HID, N);
  bn_apply_kernel<<<egrid, 256, 0, stream>>>(bufB, stats, stats + HID,
                                             bn1g, bn1b, N * HID, invN);
  // 5. conv2: agg = h1 + scatter_sum(h1)
  hipMemcpyAsync(bufA, bufB, (size_t)N * HID * sizeof(float),
                 hipMemcpyDeviceToDevice, stream);
  scatter_add_kernel<<<sgrid, 256, 0, stream>>>(bufB, srcIdx, dstIdx, bufA, E, HID);
  // 6. h2 = mlp2(agg) -> latent region of d_out
  mlp2_kernel<HID><<<mgrid, 256, 0, stream>>>(bufA, pk_w2a, b2a, pk_w2b, b2b,
                                              latent, tiles);
  // 7. latent = relu(bn2(h2)) in place
  bn_stats_kernel<<<512, 256, 0, stream>>>(latent, stats + 2 * HID,
                                           stats + 3 * HID, N);
  bn_apply_kernel<<<egrid, 256, 0, stream>>>(latent, stats + 2 * HID,
                                             stats + 3 * HID, bn2g, bn2b,
                                             N * HID, invN);
  // 8. class_out = relu(latent@mw1+mb1)@mw2+mb2
  classifier_kernel<<<mgrid, 256, 0, stream>>>(latent, pk_mw1, mb1, pk_mw2, mb2,
                                               clsout, tiles);
}
